// SpikingGCNonv_35734127903580
// MI455X (gfx1250) — matrix operand order repaired and checked by reading, not compiled, not run on hardware
//
#include <hip/hip_runtime.h>
#include <hip/hip_bf16.h>
#include <stdint.h>

// SpikingGCN for MI455X (gfx1250, wave32).
//  - Propagation: L2-resident gather/scatter with f32 atomics (x = 51MB << 192MB L2).
//  - SNN step: v_wmma_f32_16x16x32_f16; spikes generated in-register in the WMMA A layout.
//  - LIF with tau=1, v_reset=0 reduces exactly to: spike_t = (enc_t @ W.T + b >= 1).
//  - v4: LICM blocker is now an opaque *offset* (not an opaque pointer), so the
//    __shared__ base stays visible -> InferAddressSpaces keeps the fragment reads
//    on the ds_load path (not flat_load), while the t-loop still re-reads B/thr
//    from LDS each iteration and the wave stays well under 256 VGPRs.

typedef __attribute__((ext_vector_type(16))) _Float16 v16h;
typedef __attribute__((ext_vector_type(8)))  float    v8f;

#define IN_CH   128
#define OUT_CH  64
#define T_STEPS 20

__device__ __forceinline__ uint32_t pcg_hash(uint32_t v) {
    uint32_t s = v * 747796405u + 2891336453u;
    uint32_t w = ((s >> ((s >> 28u) + 4u)) ^ s) * 277803737u;
    return (w >> 22u) ^ w;
}

// ---------------- degree / normalization ----------------

__global__ void init_deg_kernel(float* __restrict__ deg, int n) {
    int i = blockIdx.x * blockDim.x + threadIdx.x;
    if (i < n) deg[i] = 1.0f;                  // self-loop contributes 1
}

__global__ void edge_deg_kernel(const long long* __restrict__ ei,
                                float* __restrict__ deg, long long e) {
    long long i = (long long)blockIdx.x * blockDim.x + threadIdx.x;
    if (i < e) atomicAdd(&deg[(int)ei[e + i]], 1.0f);   // row 1 = dst
}

__global__ void dinv_kernel(const float* __restrict__ deg,
                            float* __restrict__ dinv, int n) {
    int i = blockIdx.x * blockDim.x + threadIdx.x;
    if (i < n) dinv[i] = rsqrtf(deg[i]);       // deg >= 1 always (self loops)
}

// ---------------- propagation: out[i] = dinv[i]^2 * in[i]  (self loop) ----------------

__global__ void self_init_kernel(const float* __restrict__ in,
                                 float* __restrict__ out,
                                 const float* __restrict__ dinv, int n) {
    long long i = (long long)blockIdx.x * blockDim.x + threadIdx.x;  // float4 units
    long long total = (long long)n * (IN_CH / 4);
    if (i >= total) return;
    int node = (int)(i >> 5);                  // 32 float4 per 128-ch row
    float w = dinv[node]; w *= w;
    float4 v = reinterpret_cast<const float4*>(in)[i];
    v.x *= w; v.y *= w; v.z *= w; v.w *= w;
    reinterpret_cast<float4*>(out)[i] = v;
}

// ---------------- propagation: out[dst] += dinv[src]*dinv[dst]*in[src] ----------------
// One wave per edge: 32 lanes x float4 = 128 channels.

__global__ void edge_prop_kernel(const long long* __restrict__ ei,
                                 const float* __restrict__ dinv,
                                 const float* __restrict__ in,
                                 float* __restrict__ out, long long e) {
    const int lane = threadIdx.x & 31;
    long long edge = (long long)blockIdx.x * (blockDim.x >> 5) + (threadIdx.x >> 5);
    if (edge >= e) return;
    const long long s = ei[edge];
    const long long d = ei[e + edge];
    const float w = dinv[(int)s] * dinv[(int)d];
    const float4 xi = *reinterpret_cast<const float4*>(in + s * (long long)IN_CH + lane * 4);
    float* op = out + d * (long long)IN_CH + lane * 4;
    atomicAdd(op + 0, w * xi.x);
    atomicAdd(op + 1, w * xi.y);
    atomicAdd(op + 2, w * xi.z);
    atomicAdd(op + 3, w * xi.w);
}

// ---------------- fused Poisson-encode + GEMM(WMMA) + threshold + accumulate ----------
// One wave32 owns a 16-node tile. A: 16x32 f16 spike fragment in contiguous k order:
//   lane = M (0..15) | khalf (lane>=16); element i -> K = khalf*16 + i (same as B)
// B: W^T staged in LDS as f16; fragment element i -> K = khalf*16 + i, N = lane&15.
// C/D: f32 16x16; VGPR r -> M = r + 8*khalf, N = lane&15.
// Firing probabilities staged per-wave in LDS as u16 thresholds floor(x*65536);
// spike <=> (hash16 <= thr16). One 32-bit hash feeds two adjacent channels.

#define WAVES_PER_BLOCK 8

__global__ void __launch_bounds__(256)
snn_wmma_kernel(const float* __restrict__ xprop,
                const float* __restrict__ W,
                const float* __restrict__ b,
                float* __restrict__ out, int n) {
    __shared__ __align__(32) _Float16 sW[OUT_CH * IN_CH];               // 16 KB
    __shared__ __align__(16) uint16_t sThr[WAVES_PER_BLOCK][16 * IN_CH]; // 32 KB

    for (int i = threadIdx.x; i < OUT_CH * IN_CH; i += blockDim.x)
        sW[i] = (_Float16)W[i];
    __syncthreads();   // all threads reach this before any early exit

    const int lane  = threadIdx.x & 31;
    const int wave  = threadIdx.x >> 5;
    const int tile  = blockIdx.x * WAVES_PER_BLOCK + wave;
    const int ntile = (n + 15) >> 4;
    if (tile >= ntile) return;

    const int node_base = tile * 16;
    const int m     = lane & 15;
    const int khalf = lane >> 4;

    // ---- stage this wave's 16x128 threshold tile (wave-private: no barrier needed,
    //      same-wave LDS RAW is covered by the DScnt wait the compiler inserts).
    {
        uint32_t* dstp = reinterpret_cast<uint32_t*>(&sThr[wave][0]);
        for (int idx = lane; idx < 16 * IN_CH / 2; idx += 32) {   // u32 = 2 channels
            const int node = idx >> 6;                            // 64 u32 per row
            const int ch2  = (idx & 63) * 2;
            const int r    = min(node_base + node, n - 1);
            const float* xp = xprop + (size_t)r * IN_CH + ch2;
            const float x0 = xp[0], x1 = xp[1];
            const uint32_t t0 = min(65535u, (uint32_t)(x0 * 65536.0f));
            const uint32_t t1 = min(65535u, (uint32_t)(x1 * 65536.0f));
            dstp[idx] = t0 | (t1 << 16);
        }
    }

    float bias_n[4];
#pragma unroll
    for (int nb = 0; nb < 4; ++nb) bias_n[nb] = b[nb * 16 + m];

    const v8f vzero = {0.f, 0.f, 0.f, 0.f, 0.f, 0.f, 0.f, 0.f};
    v8f acc[4];
#pragma unroll
    for (int nb = 0; nb < 4; ++nb) acc[nb] = vzero;

    const int      row    = min(node_base + m, n - 1);
    const uint32_t rowIdx = (uint32_t)row * (uint32_t)IN_CH;
    const int wIdx = m * IN_CH + khalf * 16;   // B fragment base (+ nb/kb offsets)
    const int tIdx = m * IN_CH;                // threshold row base

    for (int t = 0; t < T_STEPS; ++t) {
        // Opaque, 16-element-aligned zero offset: the address of every LDS read
        // below formally depends on a per-iteration unknown, so LICM cannot hoist
        // the 16 B fragments (128 VGPRs) / 32 threshold dwords out of the t-loop.
        // The __shared__ bases stay visible, so these remain ds_load (not flat).
        uint32_t opq = 0;
        asm volatile("" : "+v"(opq));
        __builtin_assume((opq & 15u) == 0);    // keep ds_load_b128 merging legal

        const uint32_t tmix = (uint32_t)t * 0x9E3779B9u + 0x85EBCA6Bu;
        v8f c[4];
#pragma unroll
        for (int nb = 0; nb < 4; ++nb) c[nb] = vzero;

#pragma unroll
        for (int kb = 0; kb < 4; ++kb) {
            const int base0 = kb * 32 + khalf * 16;       // A slots i=0..7
            const int base1 = kb * 32 + khalf * 16 + 8;   // A slots i=8..15

            // 8 u16 thresholds per half, 16B aligned -> ds_load_b128
            const uint32_t* tp0 =
                reinterpret_cast<const uint32_t*>(&sThr[wave][tIdx + base0 + opq]);
            const uint32_t* tp1 =
                reinterpret_cast<const uint32_t*>(&sThr[wave][tIdx + base1 + opq]);
            uint32_t tw[8];
#pragma unroll
            for (int p = 0; p < 4; ++p) tw[p]     = tp0[p];
#pragma unroll
            for (int p = 0; p < 4; ++p) tw[4 + p] = tp1[p];

            v16h a;
#pragma unroll
            for (int p = 0; p < 8; ++p) {                 // one hash -> 2 spikes
                const int ch = (p < 4) ? (base0 + 2 * p) : (base1 + 2 * (p - 4));
                const uint32_t h = pcg_hash((rowIdx + (uint32_t)ch) ^ tmix);
                const uint32_t w = tw[p];
                a[2 * p]     = ((h & 0xffffu) <= (w & 0xffffu)) ? (_Float16)1.0f : (_Float16)0.0f;
                a[2 * p + 1] = ((h >> 16)     <= (w >> 16))     ? (_Float16)1.0f : (_Float16)0.0f;
            }

#pragma unroll
            for (int nb = 0; nb < 4; ++nb) {
                const v16h bf = *reinterpret_cast<const v16h*>(
                    &sW[wIdx + nb * 16 * IN_CH + kb * 32 + opq]);
                c[nb] = __builtin_amdgcn_wmma_f32_16x16x32_f16(
                    /*neg_a=*/false, a, /*neg_b=*/false, bf,
                    /*c_mod=*/(short)0, c[nb], /*reuse_a=*/false, /*reuse_b=*/false);
            }
        }

        // LIF with tau=1 collapses to a pure threshold on (out + bias).
#pragma unroll
        for (int nb = 0; nb < 4; ++nb)
#pragma unroll
            for (int r = 0; r < 8; ++r) {
                const float o = c[nb][r] + bias_n[nb];
                acc[nb][r] += (o >= 1.0f) ? 1.0f : 0.0f;
            }
    }

    const float inv_t = 1.0f / (float)T_STEPS;
#pragma unroll
    for (int nb = 0; nb < 4; ++nb)
#pragma unroll
        for (int r = 0; r < 8; ++r) {
            const int mr = r + (khalf ? 8 : 0);
            const int nd = node_base + mr;
            if (nd < n)
                out[(size_t)nd * OUT_CH + nb * 16 + m] = acc[nb][r] * inv_t;
        }
}

// ---------------- host ----------------

extern "C" void kernel_launch(void* const* d_in, const int* in_sizes, int n_in,
                              void* d_out, int out_size, void* d_ws, size_t ws_size,
                              hipStream_t stream) {
    const float*     x   = (const float*)d_in[0];
    const long long* ei  = (const long long*)d_in[1];   // int64 [2, E]
    const float*     W   = (const float*)d_in[2];       // [64, 128]
    const float*     b   = (const float*)d_in[3];       // [64]
    float*           out = (float*)d_out;

    const int       n = in_sizes[0] / IN_CH;
    const long long e = (long long)in_sizes[1] / 2;

    // Workspace layout: deg[N] | dinv[N] | buf0[N*128] | buf1[N*128]  (~103 MB)
    char* ws = (char*)d_ws;
    auto a512 = [](size_t v) { return (v + 511) & ~(size_t)511; };
    size_t off = 0;
    float* deg  = (float*)(ws + off); off = a512(off + (size_t)n * 4);
    float* dinv = (float*)(ws + off); off = a512(off + (size_t)n * 4);
    float* buf0 = (float*)(ws + off); off = a512(off + (size_t)n * IN_CH * 4);
    float* buf1 = (float*)(ws + off);

    const int TB = 256;

    // 1) degrees + D^{-1/2}
    init_deg_kernel<<<(n + TB - 1) / TB, TB, 0, stream>>>(deg, n);
    edge_deg_kernel<<<(unsigned)((e + TB - 1) / TB), TB, 0, stream>>>(ei, deg, e);
    dinv_kernel<<<(n + TB - 1) / TB, TB, 0, stream>>>(deg, dinv, n);

    // 2) two propagation passes (K=2), ping-pong through workspace
    const long long q4 = (long long)n * (IN_CH / 4);
    const unsigned sBlocks = (unsigned)((q4 + TB - 1) / TB);
    const unsigned eBlocks = (unsigned)((e + (TB / 32) - 1) / (TB / 32));

    self_init_kernel<<<sBlocks, TB, 0, stream>>>(x, buf0, dinv, n);
    edge_prop_kernel<<<eBlocks, TB, 0, stream>>>(ei, dinv, x, buf0, e);

    self_init_kernel<<<sBlocks, TB, 0, stream>>>(buf0, buf1, dinv, n);
    edge_prop_kernel<<<eBlocks, TB, 0, stream>>>(ei, dinv, buf0, buf1, e);

    // 3) fused SNN: Poisson encode -> WMMA GEMM -> threshold -> rate
    const int ntile = (n + 15) / 16;                   // one wave32 per 16-node tile
    snn_wmma_kernel<<<(ntile + WAVES_PER_BLOCK - 1) / WAVES_PER_BLOCK, TB, 0, stream>>>(
        buf1, W, b, out, n);
}
